// FIBERFusionBlock_1614907703841
// MI455X (gfx1250) — matrix operand order repaired and checked by reading, compiled, not run on hardware
//
#include <hip/hip_runtime.h>

// ---------------------------------------------------------------------------
// FIBER fusion block for gfx1250 (MI455X).
// bf16 WMMA GEMMs with fragment-order packed weights (no LDS in GEMM),
// wave32 cross-attention. Software pipelining via manually unrolled
// two-stage ping-pong (all fragment indices compile-time -> no spills).
// ---------------------------------------------------------------------------

typedef __attribute__((ext_vector_type(16))) __bf16        v16bf;
typedef __attribute__((ext_vector_type(8)))  float         v8f;
typedef __attribute__((ext_vector_type(4)))  float         v4f;
typedef __attribute__((ext_vector_type(4)))  unsigned int  v4u;
typedef __attribute__((ext_vector_type(2)))  unsigned int  v2u;

#define FUS    384
#define NHEADS 4
#define HD     96
#define LN_EPS 1e-5f
#define SKMAX  576

// f32 -> bf16 (round-to-nearest-even), raw bits
__device__ __forceinline__ unsigned short f2bf(float f) {
  unsigned int x = __builtin_bit_cast(unsigned int, f);
  x += 0x7FFFu + ((x >> 16) & 1u);
  return (unsigned short)(x >> 16);
}

union Frag {
  v16bf          v;
  v4u            q[2];
  unsigned int   w[8];
  unsigned short u[16];
};

__device__ __forceinline__ v4u ld16(const unsigned short* p) {
  return *(const v4u*)p;
}

// ---------------------------------------------------------------------------
// Repack W (K x N, f32 row-major) into B-fragment order bf16:
// lane (l16, hf) element i  <->  W[kblk*32 + i + 16*hf][ntile*16 + l16]
// ---------------------------------------------------------------------------
__global__ __launch_bounds__(32)
void repack_w_kernel(const float* __restrict__ W, unsigned short* __restrict__ Wp,
                     int N)
{
  const int kblk = blockIdx.x;
  const int nt   = blockIdx.y;
  const int lane = threadIdx.x;
  const int hf = lane >> 4, l16 = lane & 15;
  const int ntiles = N >> 4;
  unsigned short* dst = Wp + (((size_t)kblk * ntiles + nt) * 32 + lane) * 16;
#pragma unroll
  for (int i = 0; i < 16; i += 2) {
    unsigned int lo = f2bf(W[(size_t)(kblk * 32 + 16 * hf + i    ) * N + nt * 16 + l16]);
    unsigned int hi = f2bf(W[(size_t)(kblk * 32 + 16 * hf + i + 1) * N + nt * 16 + l16]);
    *(unsigned int*)(dst + i) = lo | (hi << 16);
  }
}

// ---------------------------------------------------------------------------
// Elementwise f32 -> bf16 (paired, packed 32-bit stores)
// ---------------------------------------------------------------------------
__global__ __launch_bounds__(256)
void cvt_bf16_kernel(const float* __restrict__ X, unsigned short* __restrict__ Y,
                     long npairs)
{
  long i = (long)blockIdx.x * 256 + threadIdx.x;
  if (i < npairs) {
    unsigned int lo = f2bf(X[2 * i]);
    unsigned int hi = f2bf(X[2 * i + 1]);
    *(unsigned int*)(Y + 2 * i) = lo | (hi << 16);
  }
}

// ---------------------------------------------------------------------------
// GEMM: C[M,N] = Abf[M,K](bf16) @ Wp(packed bf16) + bias  [+resid]
// 1 wave per block, wave tile 32x64 (2x4 of 16x16), K step 32, no LDS.
// Two-stage manual ping-pong (K/32 is always even here: 6, 12, 28).
// OM: 0 = f32, 1 = f32 + resid, 2 = bf16 row-major,
//     3 = bf16 head-transposed: Cbf[(b*N + col)*SkT + s], row = b*SkT + s
// ---------------------------------------------------------------------------
template<int OM>
__global__ __launch_bounds__(32)
void gemm_kernel(const unsigned short* __restrict__ Abf,
                 const unsigned short* __restrict__ Wp,
                 const float* __restrict__ bias,
                 const float* __restrict__ resid,
                 float* __restrict__ C,
                 unsigned short* __restrict__ Cbf,
                 int M, int N, int K, int SkT)
{
  const int lane = threadIdx.x;
  const int hf = lane >> 4, l16 = lane & 15;
  const int m0 = blockIdx.x * 32;
  const int n0 = blockIdx.y * 64;
  const int ntiles = N >> 4;
  const int nt0 = n0 >> 4;

  v8f acc[2][4];
#pragma unroll
  for (int mt = 0; mt < 2; ++mt)
#pragma unroll
    for (int nt = 0; nt < 4; ++nt) acc[mt][nt] = (v8f){};

  auto loadStage = [&](int k0, Frag (&af)[2], Frag (&bfr)[4]) {
#pragma unroll
    for (int mt = 0; mt < 2; ++mt) {
      const unsigned short* p = Abf + (size_t)(m0 + mt * 16 + l16) * K + k0 + 16 * hf;
      af[mt].q[0] = ld16(p);
      af[mt].q[1] = ld16(p + 8);
    }
    const unsigned short* wb =
        Wp + (((size_t)(k0 >> 5) * ntiles + nt0) * 32 + lane) * 16;
#pragma unroll
    for (int nt = 0; nt < 4; ++nt) {
      const unsigned short* p = wb + (size_t)nt * 512;
      bfr[nt].q[0] = ld16(p);
      bfr[nt].q[1] = ld16(p + 8);
    }
  };
  auto compute = [&](Frag (&af)[2], Frag (&bfr)[4]) {
#pragma unroll
    for (int mt = 0; mt < 2; ++mt)
#pragma unroll
      for (int nt = 0; nt < 4; ++nt)
        acc[mt][nt] = __builtin_amdgcn_wmma_f32_16x16x32_bf16(
            false, af[mt].v, false, bfr[nt].v, (short)0, acc[mt][nt],
            false, false);
  };

  Frag afA[2], bfA[4], afB[2], bfB[4];
  loadStage(0, afA, bfA);
  for (int k0 = 0; k0 < K; k0 += 64) {
    loadStage(k0 + 32, afB, bfB);      // K/32 even -> k0+32 always valid
    compute(afA, bfA);
    if (k0 + 64 < K) loadStage(k0 + 64, afA, bfA);
    compute(afB, bfB);
  }

  // epilogue
  int bidx = 0, s0 = 0;
  if constexpr (OM == 3) { bidx = m0 / SkT; s0 = m0 - bidx * SkT; }
#pragma unroll
  for (int mt = 0; mt < 2; ++mt)
#pragma unroll
    for (int nt = 0; nt < 4; ++nt)
#pragma unroll
      for (int g = 0; g < 8; ++g) {
        int row = m0 + mt * 16 + g + 8 * hf;
        int col = n0 + nt * 16 + l16;
        float v = acc[mt][nt][g] + bias[col];
        if constexpr (OM == 1) v += resid[(size_t)row * N + col];
        if constexpr (OM <= 1) {
          C[(size_t)row * N + col] = v;
        } else if constexpr (OM == 2) {
          Cbf[(size_t)row * N + col] = f2bf(v);
        } else {
          int s = s0 + mt * 16 + g + 8 * hf;
          Cbf[((size_t)bidx * N + col) * SkT + s] = f2bf(v);
        }
      }
}

// ---------------------------------------------------------------------------
// Row LayerNorm over D=384, 1 wave per row.
// MODE 0: Yf = gelu(LN); MODE 1: Ybf = bf16(LN); MODE 2: Ybf = bf16(alpha*LN)
// ---------------------------------------------------------------------------
template<int MODE>
__global__ __launch_bounds__(32)
void ln_kernel(const float* __restrict__ X, const float* __restrict__ g,
               const float* __restrict__ b, const float* __restrict__ araw,
               float* __restrict__ Yf, unsigned short* __restrict__ Ybf)
{
  const int row  = blockIdx.x;
  const int lane = threadIdx.x;
  const float* x = X + (size_t)row * FUS;
  float v[12];
  float s = 0.f;
#pragma unroll
  for (int j = 0; j < 12; ++j) { v[j] = x[lane + 32 * j]; s += v[j]; }
#pragma unroll
  for (int off = 16; off > 0; off >>= 1) s += __shfl_xor(s, off, 32);
  float mu = s * (1.0f / FUS);
  float var = 0.f;
#pragma unroll
  for (int j = 0; j < 12; ++j) { float d = v[j] - mu; var += d * d; }
#pragma unroll
  for (int off = 16; off > 0; off >>= 1) var += __shfl_xor(var, off, 32);
  float inv = rsqrtf(var * (1.0f / FUS) + LN_EPS);
  float alpha = 1.0f;
  if constexpr (MODE == 2) alpha = 0.25f * (tanhf(araw[0]) + 1.0f);
#pragma unroll
  for (int j = 0; j < 12; ++j) {
    int c = lane + 32 * j;
    float t = (v[j] - mu) * inv * g[c] + b[c];
    if constexpr (MODE == 0) {
      t = 0.5f * t * (1.0f + erff(t * 0.70710678f));
      Yf[(size_t)row * FUS + c] = t;
    } else if constexpr (MODE == 1) {
      Ybf[(size_t)row * FUS + c] = f2bf(t);
    } else {
      Ybf[(size_t)row * FUS + c] = f2bf(alpha * t);
    }
  }
}

// ---------------------------------------------------------------------------
// Cross-attention: one wave per (b, h, 16-query tile).
// Q/K bf16 row-major, V bf16 head-transposed: VT[(b*FUS + h*96+d)*Sk + s].
// Two-stage ping-pong QK and PV loops; vectorized softmax.
// ---------------------------------------------------------------------------
template<bool HAS_MASK>
__global__ __launch_bounds__(32)
void attn_kernel(const unsigned short* __restrict__ Qbf,
                 const unsigned short* __restrict__ Kbf,
                 const unsigned short* __restrict__ VT,
                 const int* __restrict__ mask,
                 unsigned short* __restrict__ Outbf, int Sq, int Sk)
{
  __shared__ __align__(16) float          sc[16 * SKMAX];
  __shared__ __align__(16) unsigned short pb[16 * SKMAX];

  const int lane = threadIdx.x;
  const int hf = lane >> 4, l16 = lane & 15;
  const int q0 = blockIdx.x * 16;
  const int h  = blockIdx.y;
  const int bb = blockIdx.z;
  const size_t qrow0 = (size_t)bb * Sq + q0;
  const size_t krow0 = (size_t)bb * Sk;
  const int hc = h * HD;
  const float scale = rsqrtf((float)HD);

  // Q fragments (A layout): 3 chunks of K32, each = 2 x 16B loads
  Frag qf[3];
#pragma unroll
  for (int c = 0; c < 3; ++c) {
    const unsigned short* p = Qbf + (qrow0 + l16) * FUS + hc + c * 32 + 16 * hf;
    qf[c].q[0] = ld16(p);
    qf[c].q[1] = ld16(p + 8);
  }

  // ---- scores = clip(QK^T * scale) (+mask); Sk/16 even -> 2-stage ----
  auto loadK = [&](int n0t, Frag (&kf)[3]) {
#pragma unroll
    for (int c = 0; c < 3; ++c) {
      const unsigned short* p =
          Kbf + (krow0 + n0t + l16) * FUS + hc + c * 32 + 16 * hf;
      kf[c].q[0] = ld16(p);
      kf[c].q[1] = ld16(p + 8);
    }
  };
  auto scoreTile = [&](int n0t, Frag (&kf)[3]) {
    v8f a = (v8f){};
#pragma unroll
    for (int c = 0; c < 3; ++c)
      a = __builtin_amdgcn_wmma_f32_16x16x32_bf16(
          false, qf[c].v, false, kf[c].v, (short)0, a, false, false);
    int key = n0t + l16;
    bool km = false;
    if constexpr (HAS_MASK) km = (mask[(size_t)bb * Sk + key] == 0);
#pragma unroll
    for (int g = 0; g < 8; ++g) {
      float v = a[g] * scale;
      v = fminf(fmaxf(v, -50.f), 50.f);
      if (km) v = -__builtin_inff();
      sc[(g + 8 * hf) * SKMAX + key] = v;
    }
  };
  {
    Frag kfA[3], kfB[3];
    loadK(0, kfA);
    for (int n0t = 0; n0t < Sk; n0t += 32) {
      loadK(n0t + 16, kfB);
      scoreTile(n0t, kfA);
      if (n0t + 32 < Sk) loadK(n0t + 32, kfA);
      scoreTile(n0t + 16, kfB);
    }
  }
  __syncthreads();

  // ---- softmax: lanes (r, r+16) own contiguous halves of row r ----
  {
    const int r = l16;
    const int hlen = Sk >> 1;
    float* srow = sc + r * SKMAX + hf * hlen;
    float mx = -__builtin_inff();
    for (int c = 0; c < hlen; c += 4) {
      v4f t = *(const v4f*)(srow + c);
      mx = fmaxf(mx, fmaxf(fmaxf(t[0], t[1]), fmaxf(t[2], t[3])));
    }
    mx = fmaxf(mx, __shfl_xor(mx, 16, 32));
    const bool dead = (mx == -__builtin_inff());
    float sum = 0.f;
    for (int c = 0; c < hlen; c += 4) {
      v4f t = *(const v4f*)(srow + c);
      v4f e;
#pragma unroll
      for (int j = 0; j < 4; ++j) e[j] = dead ? 0.f : __expf(t[j] - mx);
      *(v4f*)(srow + c) = e;
      sum += (e[0] + e[1]) + (e[2] + e[3]);
    }
    sum += __shfl_xor(sum, 16, 32);
    const float invs = (sum > 0.f) ? (1.0f / sum) : 0.f;
    unsigned short* prow = pb + r * SKMAX + hf * hlen;
    for (int c = 0; c < hlen; c += 4) {
      v4f t = *(const v4f*)(srow + c);
      v2u o;
      o[0] = (unsigned int)f2bf(t[0] * invs) | ((unsigned int)f2bf(t[1] * invs) << 16);
      o[1] = (unsigned int)f2bf(t[2] * invs) | ((unsigned int)f2bf(t[3] * invs) << 16);
      *(v2u*)(prow + c) = o;
    }
  }
  __syncthreads();

  // ---- out = P @ V; Sk/32 even -> 2-stage ping-pong ----
  v8f oacc[6];
#pragma unroll
  for (int nt = 0; nt < 6; ++nt) oacc[nt] = (v8f){};
  auto loadPV = [&](int kc, Frag& pf, Frag (&vfr)[6]) {
    const unsigned short* pp = pb + l16 * SKMAX + kc + 16 * hf;
    pf.q[0] = ld16(pp);
    pf.q[1] = ld16(pp + 8);
#pragma unroll
    for (int nt = 0; nt < 6; ++nt) {
      const unsigned short* p =
          VT + ((size_t)bb * FUS + hc + nt * 16 + l16) * Sk + kc + 16 * hf;
      vfr[nt].q[0] = ld16(p);
      vfr[nt].q[1] = ld16(p + 8);
    }
  };
  auto computePV = [&](Frag& pf, Frag (&vfr)[6]) {
#pragma unroll
    for (int nt = 0; nt < 6; ++nt)
      oacc[nt] = __builtin_amdgcn_wmma_f32_16x16x32_bf16(
          false, pf.v, false, vfr[nt].v, (short)0, oacc[nt], false, false);
  };
  {
    Frag pfA, vfA[6], pfB, vfB[6];
    loadPV(0, pfA, vfA);
    for (int kc = 0; kc < Sk; kc += 64) {
      loadPV(kc + 32, pfB, vfB);
      computePV(pfA, vfA);
      if (kc + 64 < Sk) loadPV(kc + 64, pfA, vfA);
      computePV(pfB, vfB);
    }
  }
#pragma unroll
  for (int nt = 0; nt < 6; ++nt)
#pragma unroll
    for (int g = 0; g < 8; ++g)
      Outbf[(qrow0 + g + 8 * hf) * FUS + hc + nt * 16 + l16] = f2bf(oacc[nt][g]);
}

// ---------------------------------------------------------------------------
// Host orchestration
// ---------------------------------------------------------------------------
extern "C" void kernel_launch(void* const* d_in, const int* in_sizes, int n_in,
                              void* d_out, int out_size, void* d_ws, size_t ws_size,
                              hipStream_t stream)
{
  (void)in_sizes; (void)n_in; (void)out_size; (void)ws_size;

  const int B = 32, P = 576, S = 512;
  const int VIS = 192, TXT = 896;
  const int MV = B * P;   // 18432
  const int MT = B * S;   // 16384

  const float* vision = (const float*)d_in[0];
  const float* text   = (const float*)d_in[1];
  const int*   tmask  = (const int*)d_in[2];
  const float* vup_w = (const float*)d_in[3],  * vup_b  = (const float*)d_in[4];
  const float* vup_g = (const float*)d_in[5],  * vup_bb = (const float*)d_in[6];
  const float* tdn_w = (const float*)d_in[7],  * tdn_b  = (const float*)d_in[8];
  const float* tdn_g = (const float*)d_in[9],  * tdn_bb = (const float*)d_in[10];
  const float* vdn_w = (const float*)d_in[11], * vdn_b  = (const float*)d_in[12];
  const float* tup_w = (const float*)d_in[13], * tup_b  = (const float*)d_in[14];
  const int I2T = 15, T2I = 30;
  auto prm = [&](int base, int j) { return (const float*)d_in[base + j]; };

  // ---------------- workspace layout ----------------
  char* ws = (char*)d_ws;
  size_t off = 0;
  auto alloc = [&](size_t bytes) { void* p = ws + off; off += (bytes + 255) & ~(size_t)255; return p; };

  unsigned short* wp_vup = (unsigned short*)alloc((size_t)192 * 384 * 2);
  unsigned short* wp_tdn = (unsigned short*)alloc((size_t)896 * 384 * 2);
  unsigned short* wp_qkv[2][4];
  for (int a = 0; a < 2; ++a)
    for (int j = 0; j < 4; ++j)
      wp_qkv[a][j] = (unsigned short*)alloc((size_t)384 * 384 * 2);
  unsigned short* wp_vdn = (unsigned short*)alloc((size_t)384 * 192 * 2);
  unsigned short* wp_tup = (unsigned short*)alloc((size_t)384 * 896 * 2);

  unsigned short* vis_bf = (unsigned short*)alloc((size_t)MV * VIS * 2);
  unsigned short* txt_bf = (unsigned short*)alloc((size_t)MT * TXT * 2);
  float*          VF     = (float*)alloc((size_t)MV * FUS * 4);
  float*          TF     = (float*)alloc((size_t)MT * FUS * 4);
  unsigned short* SN     = (unsigned short*)alloc((size_t)MV * FUS * 2);
  unsigned short* TN     = (unsigned short*)alloc((size_t)MV * FUS * 2);
  unsigned short* QB     = (unsigned short*)alloc((size_t)MV * FUS * 2);
  unsigned short* KB     = (unsigned short*)alloc((size_t)MV * FUS * 2);
  unsigned short* VT     = (unsigned short*)alloc((size_t)MV * FUS * 2);
  unsigned short* AO     = (unsigned short*)alloc((size_t)MV * FUS * 2);
  float*          OF     = (float*)alloc((size_t)MV * FUS * 4);
  unsigned short* DB     = (unsigned short*)alloc((size_t)MV * FUS * 2);

  float* out_v = (float*)d_out;
  float* out_t = out_v + (size_t)MV * VIS;

  dim3 b32(32), b256(256);

  // ---------------- weight repack ----------------
  repack_w_kernel<<<dim3(192 / 32, 384 / 16), b32, 0, stream>>>(vup_w, wp_vup, 384);
  repack_w_kernel<<<dim3(896 / 32, 384 / 16), b32, 0, stream>>>(tdn_w, wp_tdn, 384);
  for (int a = 0; a < 2; ++a) {
    int base = a == 0 ? I2T : T2I;
    for (int j = 0; j < 4; ++j)
      repack_w_kernel<<<dim3(384 / 32, 384 / 16), b32, 0, stream>>>(
          prm(base, 2 * j), wp_qkv[a][j], 384);
  }
  repack_w_kernel<<<dim3(384 / 32, 192 / 16), b32, 0, stream>>>(vdn_w, wp_vdn, 192);
  repack_w_kernel<<<dim3(384 / 32, 896 / 16), b32, 0, stream>>>(tup_w, wp_tup, 896);

  // ---------------- input conversion ----------------
  {
    long np = (long)MV * VIS / 2;
    cvt_bf16_kernel<<<(np + 255) / 256, b256, 0, stream>>>(vision, vis_bf, np);
    np = (long)MT * TXT / 2;
    cvt_bf16_kernel<<<(np + 255) / 256, b256, 0, stream>>>(text, txt_bf, np);
  }

  // ---------------- vf / tf ----------------
  gemm_kernel<0><<<dim3(MV / 32, FUS / 64), b32, 0, stream>>>(
      vis_bf, wp_vup, vup_b, nullptr, VF, nullptr, MV, FUS, VIS, 0);
  ln_kernel<0><<<MV, b32, 0, stream>>>(VF, vup_g, vup_bb, nullptr, VF, nullptr);
  gemm_kernel<0><<<dim3(MT / 32, FUS / 64), b32, 0, stream>>>(
      txt_bf, wp_tdn, tdn_b, nullptr, TF, nullptr, MT, FUS, TXT, 0);
  ln_kernel<0><<<MT, b32, 0, stream>>>(TF, tdn_g, tdn_bb, nullptr, TF, nullptr);

  // ================= i2t: src=vf (MV rows), tgt=tf (MT rows), mask =========
  ln_kernel<1><<<MV, b32, 0, stream>>>(VF, prm(I2T, 8), prm(I2T, 9), nullptr, nullptr, SN);
  ln_kernel<1><<<MT, b32, 0, stream>>>(TF, prm(I2T, 10), prm(I2T, 11), nullptr, nullptr, TN);
  gemm_kernel<2><<<dim3(MV / 32, FUS / 64), b32, 0, stream>>>(
      SN, wp_qkv[0][0], prm(I2T, 1), nullptr, nullptr, QB, MV, FUS, FUS, 0);
  gemm_kernel<2><<<dim3(MT / 32, FUS / 64), b32, 0, stream>>>(
      TN, wp_qkv[0][1], prm(I2T, 3), nullptr, nullptr, KB, MT, FUS, FUS, 0);
  gemm_kernel<3><<<dim3(MT / 32, FUS / 64), b32, 0, stream>>>(
      TN, wp_qkv[0][2], prm(I2T, 5), nullptr, nullptr, VT, MT, FUS, FUS, S);
  attn_kernel<true><<<dim3(P / 16, NHEADS, B), b32, 0, stream>>>(
      QB, KB, VT, tmask, AO, P, S);
  gemm_kernel<0><<<dim3(MV / 32, FUS / 64), b32, 0, stream>>>(
      AO, wp_qkv[0][3], prm(I2T, 7), nullptr, OF, nullptr, MV, FUS, FUS, 0);
  ln_kernel<2><<<MV, b32, 0, stream>>>(
      OF, prm(I2T, 12), prm(I2T, 13), prm(I2T, 14), nullptr, DB); // = v_fused - vf
  gemm_kernel<1><<<dim3(MV / 32, VIS / 64), b32, 0, stream>>>(
      DB, wp_vdn, vdn_b, vision, out_v, nullptr, MV, VIS, FUS, 0);

  // ================= t2i: src=tf (MT rows), tgt=vf (MV rows), no mask ======
  ln_kernel<1><<<MT, b32, 0, stream>>>(TF, prm(T2I, 8), prm(T2I, 9), nullptr, nullptr, SN);
  ln_kernel<1><<<MV, b32, 0, stream>>>(VF, prm(T2I, 10), prm(T2I, 11), nullptr, nullptr, TN);
  gemm_kernel<2><<<dim3(MT / 32, FUS / 64), b32, 0, stream>>>(
      SN, wp_qkv[1][0], prm(T2I, 1), nullptr, nullptr, QB, MT, FUS, FUS, 0);
  gemm_kernel<2><<<dim3(MV / 32, FUS / 64), b32, 0, stream>>>(
      TN, wp_qkv[1][1], prm(T2I, 3), nullptr, nullptr, KB, MV, FUS, FUS, 0);
  gemm_kernel<3><<<dim3(MV / 32, FUS / 64), b32, 0, stream>>>(
      TN, wp_qkv[1][2], prm(T2I, 5), nullptr, nullptr, VT, MV, FUS, FUS, P);
  attn_kernel<false><<<dim3(S / 16, NHEADS, B), b32, 0, stream>>>(
      QB, KB, VT, nullptr, AO, S, P);
  gemm_kernel<0><<<dim3(MT / 32, FUS / 64), b32, 0, stream>>>(
      AO, wp_qkv[1][3], prm(T2I, 7), nullptr, OF, nullptr, MT, FUS, FUS, 0);
  ln_kernel<2><<<MT, b32, 0, stream>>>(
      OF, prm(T2I, 12), prm(T2I, 13), prm(T2I, 14), nullptr, DB); // = t_fused - tf
  gemm_kernel<1><<<dim3(MT / 32, TXT / 64), b32, 0, stream>>>(
      DB, wp_tup, tup_b, text, out_t, nullptr, MT, TXT, FUS, 0);
}
